// SAGE_2508260901292
// MI455X (gfx1250) — matrix operand order repaired and checked by reading, not compiled, not run on hardware
//
#include <hip/hip_runtime.h>
#include <hip/hip_bf16.h>
#include <math.h>

typedef _Float16 half_t;
typedef __attribute__((ext_vector_type(8)))  _Float16 v8h;
typedef __attribute__((ext_vector_type(16))) _Float16 v16h;
typedef __attribute__((ext_vector_type(8)))  float    v8f;

#define N_NODES  50000
#define N_EDGES  600000
#define N_GRAPHS 256
#define D        128     // N_FL1 == N_MPL == N_FL2 == 128
#define D2       64      // N_FL3

// ---------------------------------------------------------------------------
// Degree: count edges per dst, then invert (deg_inv = 1/max(deg,1))
// ---------------------------------------------------------------------------
__global__ void sage_deg_count(const int* __restrict__ dst, float* __restrict__ deg, int n_edges) {
    int e = blockIdx.x * blockDim.x + threadIdx.x;
    if (e < n_edges) atomicAdd(&deg[dst[e]], 1.0f);
}

__global__ void sage_deg_invert(float* __restrict__ deg, int n_nodes) {
    int n = blockIdx.x * blockDim.x + threadIdx.x;
    if (n < n_nodes) deg[n] = 1.0f / fmaxf(deg[n], 1.0f);
}

// ---------------------------------------------------------------------------
// Pre-MLP: h = relu(x @ W_pre + b_pre); x:[N,5], W:[5,128] -> f16 features
// ---------------------------------------------------------------------------
__global__ __launch_bounds__(D) void sage_pre_mlp(const float* __restrict__ x,
                                                  const float* __restrict__ W,
                                                  const float* __restrict__ b,
                                                  half_t* __restrict__ h) {
    __shared__ float xs[5];
    int n = blockIdx.x;
    int j = threadIdx.x;
    if (j < 5) xs[j] = x[n * 5 + j];
    __syncthreads();
    float s = b[j];
#pragma unroll
    for (int k = 0; k < 5; ++k) s += xs[k] * W[k * D + j];
    h[n * D + j] = (half_t)fmaxf(s, 0.0f);
}

// ---------------------------------------------------------------------------
// Pack [W_l;W_r] (each [128,128] f32 row-major) into transposed f16
// Wt[n][k] with n in [0,128), k in [0,256): k<128 -> W_l[k][n], else W_r.
// Makes every WMMA B-fragment load two contiguous 16B runs per lane.
// ---------------------------------------------------------------------------
__global__ void sage_pack_weights(const float* __restrict__ Wl,
                                  const float* __restrict__ Wr,
                                  half_t* __restrict__ Wt) {
    int t = blockIdx.x * blockDim.x + threadIdx.x;   // 128*256 threads
    if (t >= D * 2 * D) return;
    int n = t >> 8;          // output column
    int k = t & 255;         // combined K index
    float v = (k < D) ? Wl[k * D + n] : Wr[(k - D) * D + n];
    Wt[t] = (half_t)v;
}

// ---------------------------------------------------------------------------
// Edge scatter: agg[dst] += h[src] (f32 atomics, 8 feats per thread)
// ---------------------------------------------------------------------------
__global__ void sage_scatter(const half_t* __restrict__ h,
                             const int* __restrict__ src,
                             const int* __restrict__ dst,
                             float* __restrict__ agg, int n_edges) {
    int t = blockIdx.x * blockDim.x + threadIdx.x;
    int e = t >> 4;               // 16 chunks of 8 features each
    if (e >= n_edges) return;
    int c = (t & 15) * 8;
    int s = src[e], d = dst[e];
    v8h hv = *(const v8h*)(h + s * D + c);
    float* out = agg + d * D + c;
#pragma unroll
    for (int i = 0; i < 8; ++i) atomicAdd(out + i, (float)hv[i]);
}

// ---------------------------------------------------------------------------
// Scale by deg_inv and convert to f16 for the WMMA A operand
// ---------------------------------------------------------------------------
__global__ void sage_scale_agg(const float* __restrict__ agg,
                               const float* __restrict__ deg_inv,
                               half_t* __restrict__ aggh, int n_nodes) {
    int t = blockIdx.x * blockDim.x + threadIdx.x;
    int n = t >> 7;
    if (n >= n_nodes) return;
    aggh[t] = (half_t)(agg[t] * deg_inv[n]);
}

// ---------------------------------------------------------------------------
// Fused SAGE GEMM: out = relu(L2norm(agg@W_l + b + h@W_r))
// One wave owns a 16(M) x 128(N) stripe; K = 256 (phase 0: agg, phase 1: h).
// 8x v_wmma_f32_16x16x32_f16 accumulators; norm via shfl_xor over lane halves.
// ---------------------------------------------------------------------------
__global__ __launch_bounds__(128) void sage_conv_wmma(const half_t* __restrict__ Aagg,
                                                      const half_t* __restrict__ Ah,
                                                      const half_t* __restrict__ Wt,
                                                      const float* __restrict__ bias,
                                                      half_t* __restrict__ out, int M) {
    int wave = threadIdx.x >> 5;
    int lane = threadIdx.x & 31;
    int row_base = (blockIdx.x * 4 + wave) * 16;
    if (row_base >= M) return;                // wave-uniform: EXEC stays all-ones
    int laneN  = lane & 15;
    int laneHi = lane >> 4;

    v8f acc[8];
#pragma unroll
    for (int t = 0; t < 8; ++t) {
        v8f z = {0.f, 0.f, 0.f, 0.f, 0.f, 0.f, 0.f, 0.f};
        acc[t] = z;
    }

#pragma unroll
    for (int phase = 0; phase < 2; ++phase) {
        const half_t* A = phase ? Ah : Aagg;
        const half_t* arow = A + (size_t)(row_base + laneN) * D;
#pragma unroll
        for (int kc = 0; kc < D; kc += 32) {
            // A fragment (16x32 f16): lane holds row=row_base+laneN,
            // K = kc + laneHi*8 + {0..7} and +16..23
            int kA = kc + laneHi * 8;
            v8h alo = *(const v8h*)(arow + kA);
            v8h ahi = *(const v8h*)(arow + kA + 16);
            v16h a = __builtin_shufflevector(alo, ahi,
                0, 1, 2, 3, 4, 5, 6, 7, 8, 9, 10, 11, 12, 13, 14, 15);
            int kg = phase * D + kc + laneHi * 8;
#pragma unroll
            for (int t = 0; t < 8; ++t) {
                // B fragment (32x16 f16): lane holds col n = t*16+laneN,
                // K = kc + laneHi*8 + {0..7} and +16..23 from the transposed weight row
                const half_t* brow = Wt + (size_t)(t * 16 + laneN) * (2 * D) + kg;
                v8h blo = *(const v8h*)(brow);
                v8h bhi = *(const v8h*)(brow + 16);
                v16h b = __builtin_shufflevector(blo, bhi,
                    0, 1, 2, 3, 4, 5, 6, 7, 8, 9, 10, 11, 12, 13, 14, 15);
                acc[t] = __builtin_amdgcn_wmma_f32_16x16x32_f16(
                    false, a, false, b, (short)0, acc[t], false, false);
            }
        }
    }

    // bias (per output column)
#pragma unroll
    for (int t = 0; t < 8; ++t) {
        float bc = bias[t * 16 + laneN];
#pragma unroll
        for (int r = 0; r < 8; ++r) acc[t][r] += bc;
    }

    // per-row L2 normalize (eps=1e-12), relu, store f16
#pragma unroll
    for (int r = 0; r < 8; ++r) {
        float s = 0.f;
#pragma unroll
        for (int t = 0; t < 8; ++t) s += acc[t][r] * acc[t][r];
        // reduce across the 16 lanes that share this row (stay within half)
        s += __shfl_xor(s, 1, 32);
        s += __shfl_xor(s, 2, 32);
        s += __shfl_xor(s, 4, 32);
        s += __shfl_xor(s, 8, 32);
        float inv = 1.0f / fmaxf(sqrtf(s), 1e-12f);
        int row = row_base + laneHi * 8 + r;   // C layout: lanes>=16 hold rows 8..15
        half_t* orow = out + (size_t)row * D;
#pragma unroll
        for (int t = 0; t < 8; ++t) {
            float v = fmaxf(acc[t][r] * inv, 0.0f);
            orow[t * 16 + laneN] = (half_t)v;
        }
    }
}

// ---------------------------------------------------------------------------
// Global mean pool accumulation
// ---------------------------------------------------------------------------
__global__ void sage_pool(const half_t* __restrict__ h,
                          const int* __restrict__ batch,
                          float* __restrict__ gsum,
                          float* __restrict__ gcnt, int n_nodes) {
    int t = blockIdx.x * blockDim.x + threadIdx.x;
    int n = t >> 7;
    if (n >= n_nodes) return;
    int j = t & 127;
    int g = batch[n];
    atomicAdd(&gsum[g * D + j], (float)h[t]);
    if (j == 0) atomicAdd(&gcnt[g], 1.0f);
}

// ---------------------------------------------------------------------------
// Head MLP: g -> relu(@W_p1+b) -> relu(@W_p2+b) -> @W_out+b  (256 graphs, f32)
// ---------------------------------------------------------------------------
__global__ __launch_bounds__(D) void sage_head(const float* __restrict__ gsum,
                                               const float* __restrict__ gcnt,
                                               const float* __restrict__ Wp1,
                                               const float* __restrict__ bp1,
                                               const float* __restrict__ Wp2,
                                               const float* __restrict__ bp2,
                                               const float* __restrict__ Wout,
                                               const float* __restrict__ bout,
                                               float* __restrict__ out) {
    __shared__ float s0[D], s1[D], s2[D2];
    int g = blockIdx.x;
    int j = threadIdx.x;
    float cnt = fmaxf(gcnt[g], 1.0f);
    s0[j] = gsum[g * D + j] / cnt;
    __syncthreads();
    float a1 = bp1[j];
    for (int k = 0; k < D; ++k) a1 += s0[k] * Wp1[k * D + j];
    s1[j] = fmaxf(a1, 0.0f);
    __syncthreads();
    if (j < D2) {
        float a2 = bp2[j];
        for (int k = 0; k < D; ++k) a2 += s1[k] * Wp2[k * D2 + j];
        s2[j] = fmaxf(a2, 0.0f);
    }
    __syncthreads();
    if (j == 0) {
        float o = bout[0];
        for (int k = 0; k < D2; ++k) o += s2[k] * Wout[k];
        out[g] = o;
    }
}

// ---------------------------------------------------------------------------
extern "C" void kernel_launch(void* const* d_in, const int* in_sizes, int n_in,
                              void* d_out, int out_size, void* d_ws, size_t ws_size,
                              hipStream_t stream) {
    (void)in_sizes; (void)n_in; (void)out_size; (void)ws_size;

    const float* x     = (const float*)d_in[0];
    const int*   ei    = (const int*)  d_in[1];   // [2, N_EDGES] row-major
    const int*   batch = (const int*)  d_in[2];
    const float* W_pre = (const float*)d_in[3];
    const float* b_pre = (const float*)d_in[4];
    const float* W1_l  = (const float*)d_in[5];
    const float* b1    = (const float*)d_in[6];
    const float* W1_r  = (const float*)d_in[7];
    const float* W2_l  = (const float*)d_in[8];
    const float* b2    = (const float*)d_in[9];
    const float* W2_r  = (const float*)d_in[10];
    const float* W_p1  = (const float*)d_in[11];
    const float* b_p1  = (const float*)d_in[12];
    const float* W_p2  = (const float*)d_in[13];
    const float* b_p2  = (const float*)d_in[14];
    const float* W_out = (const float*)d_in[15];
    const float* b_out = (const float*)d_in[16];
    float* out = (float*)d_out;

    const int* src = ei;
    const int* dst = ei + N_EDGES;

    // workspace carve-up (256B aligned)
    char* ws = (char*)d_ws;
    size_t off = 0;
    auto carve = [&](size_t bytes) -> void* {
        off = (off + 255) & ~(size_t)255;
        void* p = ws + off;
        off += bytes;
        return p;
    };
    float*  deg   = (float*) carve((size_t)N_NODES * 4);
    half_t* hA    = (half_t*)carve((size_t)N_NODES * D * 2);
    half_t* hB    = (half_t*)carve((size_t)N_NODES * D * 2);
    float*  aggF  = (float*) carve((size_t)N_NODES * D * 4);
    half_t* aggH  = (half_t*)carve((size_t)N_NODES * D * 2);
    half_t* Wt1   = (half_t*)carve((size_t)D * 2 * D * 2);
    half_t* Wt2   = (half_t*)carve((size_t)D * 2 * D * 2);
    float*  gsum  = (float*) carve((size_t)N_GRAPHS * D * 4);
    float*  gcnt  = (float*) carve((size_t)N_GRAPHS * 4);

    const int ntile   = N_NODES / 16;                 // 3125 (exact)
    const int gemmGrd = (ntile + 3) / 4;              // 4 waves/block
    const int elemGrd = (N_NODES * D + 255) / 256;    // per-element kernels
    const int scatGrd = (N_EDGES * 16 + 255) / 256;

    // degrees
    hipMemsetAsync(deg, 0, (size_t)N_NODES * 4, stream);
    sage_deg_count <<<(N_EDGES + 255) / 256, 256, 0, stream>>>(dst, deg, N_EDGES);
    sage_deg_invert<<<(N_NODES + 255) / 256, 256, 0, stream>>>(deg, N_NODES);

    // pre-MLP + weight packing
    sage_pre_mlp<<<N_NODES, D, 0, stream>>>(x, W_pre, b_pre, hA);
    sage_pack_weights<<<(D * 2 * D + 255) / 256, 256, 0, stream>>>(W1_l, W1_r, Wt1);
    sage_pack_weights<<<(D * 2 * D + 255) / 256, 256, 0, stream>>>(W2_l, W2_r, Wt2);

    // conv 1
    hipMemsetAsync(aggF, 0, (size_t)N_NODES * D * 4, stream);
    sage_scatter   <<<scatGrd, 256, 0, stream>>>(hA, src, dst, aggF, N_EDGES);
    sage_scale_agg <<<elemGrd, 256, 0, stream>>>(aggF, deg, aggH, N_NODES);
    sage_conv_wmma <<<gemmGrd, 128, 0, stream>>>(aggH, hA, Wt1, b1, hB, N_NODES);

    // conv 2
    hipMemsetAsync(aggF, 0, (size_t)N_NODES * D * 4, stream);
    sage_scatter   <<<scatGrd, 256, 0, stream>>>(hB, src, dst, aggF, N_EDGES);
    sage_scale_agg <<<elemGrd, 256, 0, stream>>>(aggF, deg, aggH, N_NODES);
    sage_conv_wmma <<<gemmGrd, 128, 0, stream>>>(aggH, hB, Wt2, b2, hA, N_NODES);

    // pooling + head
    hipMemsetAsync(gsum, 0, (size_t)N_GRAPHS * D * 4, stream);
    hipMemsetAsync(gcnt, 0, (size_t)N_GRAPHS * 4, stream);
    sage_pool<<<elemGrd, 256, 0, stream>>>(hA, batch, gsum, gcnt, N_NODES);
    sage_head<<<N_GRAPHS, D, 0, stream>>>(gsum, gcnt, W_p1, b_p1, W_p2, b_p2,
                                          W_out, b_out, out);
}
